// TransformerTransferGSCPredictor_58909771432904
// MI455X (gfx1250) — matrix operand order repaired and checked by reading, not compiled, not run on hardware
//
#include <hip/hip_runtime.h>

typedef __attribute__((ext_vector_type(16))) __bf16 v16bf;
typedef __attribute__((ext_vector_type(8)))  float  v8f;

#define N_OBJ   4096
#define N_RELP  16384
#define HID     512
#define POOL    4096
#define KRANK   100
#define NREL    51
#define NOBJC   151
#define CATW    4296   // 2*KRANK + POOL
#define LDSS    72     // 64 + 8 pad (bf16 units)

static __device__ __forceinline__ unsigned short f2bf(float f) {
    union { float f; unsigned u; } x; x.f = f;
    unsigned r = x.u + 0x7FFFu + ((x.u >> 16) & 1u);
    return (unsigned short)(r >> 16);
}

// ---------------------------------------------------------------------------
// Tiled bf16 WMMA GEMM: C[M,N] = epilogue(A[M,K] @ Bt[N,K]^T + bias)
// 256 threads = 8 waves; block tile 128x128; K tile 64 (two 16x16x32 WMMA steps)
// Staging is software-pipelined: tile t+1 global loads are issued before the
// WMMA work of tile t, so HBM latency overlaps matrix math.
// epi: 0=bias  1=bias+relu  2=relu(aux*(acc+bias))  3=relu(acc * (*scale))
//      4=C += acc + bias (f32 C only)
// ---------------------------------------------------------------------------
__global__ __launch_bounds__(256) void gemm_bf16_wmma(
    const unsigned short* __restrict__ A,  int ldA,
    const unsigned short* __restrict__ Bt, int ldB,
    void* __restrict__ Cv, int ldC, int c_bf16,
    int M, int N, int K,
    const float* __restrict__ bias, int epi,
    const float* __restrict__ aux, int ldAux,
    const float* __restrict__ scale_ptr)
{
    __shared__ unsigned short Asl[128 * LDSS];
    __shared__ unsigned short Bsl[128 * LDSS];

    const int tid  = threadIdx.x;
    const int lane = tid & 31;
    const int wave = tid >> 5;
    const int m0 = blockIdx.y * 128;
    const int n0 = blockIdx.x * 128;
    const int wm = (wave >> 1) * 32;   // 0,32,64,96
    const int wn = (wave & 1)  * 64;   // 0,64
    const int hlf = lane >> 4;
    const int l16 = lane & 15;

    v8f acc[2][4];
    const v8f vzero = {0.f,0.f,0.f,0.f,0.f,0.f,0.f,0.f};
    #pragma unroll
    for (int i = 0; i < 2; ++i)
        #pragma unroll
        for (int j = 0; j < 4; ++j) acc[i][j] = vzero;

    const int lrow = tid >> 1;          // 0..127
    const int lseg = (tid & 1) * 32;    // 0 or 32 (bf16 elems)
    const int gmA = m0 + lrow;
    const int gnB = n0 + lrow;

    // block-uniform fast-path eligibility (interior in M and N, b128-aligned rows)
    const bool interior = (m0 + 128 <= M) && (n0 + 128 <= N) &&
                          ((ldA & 7) == 0) && ((ldB & 7) == 0);

    const unsigned short* aRow = A  + (size_t)gmA * ldA + lseg;
    const unsigned short* bRow = Bt + (size_t)gnB * ldB + lseg;
    unsigned short* aDst = &Asl[lrow * LDSS + lseg];
    unsigned short* bDst = &Bsl[lrow * LDSS + lseg];

    // load one 64-wide K-tile worth of per-thread data into registers
    auto load_tile = [&](int k0, uint4* ra, uint4* rb) {
        if (interior && (k0 + 64 <= K)) {
            const uint4* sa = (const uint4*)(aRow + k0);
            const uint4* sb = (const uint4*)(bRow + k0);
            #pragma unroll
            for (int c = 0; c < 4; ++c) ra[c] = sa[c];
            #pragma unroll
            for (int c = 0; c < 4; ++c) rb[c] = sb[c];
        } else {
            #pragma unroll
            for (int c = 0; c < 4; ++c) {
                unsigned w[4];
                #pragma unroll
                for (int e = 0; e < 4; ++e) {
                    int kk = k0 + lseg + 8 * c + 2 * e;
                    unsigned lo = (gmA < M && kk     < K) ? (unsigned)A[(size_t)gmA * ldA + kk]     : 0u;
                    unsigned hi = (gmA < M && kk + 1 < K) ? (unsigned)A[(size_t)gmA * ldA + kk + 1] : 0u;
                    w[e] = lo | (hi << 16);
                }
                ra[c].x = w[0]; ra[c].y = w[1]; ra[c].z = w[2]; ra[c].w = w[3];
            }
            #pragma unroll
            for (int c = 0; c < 4; ++c) {
                unsigned w[4];
                #pragma unroll
                for (int e = 0; e < 4; ++e) {
                    int kk = k0 + lseg + 8 * c + 2 * e;
                    unsigned lo = (gnB < N && kk     < K) ? (unsigned)Bt[(size_t)gnB * ldB + kk]     : 0u;
                    unsigned hi = (gnB < N && kk + 1 < K) ? (unsigned)Bt[(size_t)gnB * ldB + kk + 1] : 0u;
                    w[e] = lo | (hi << 16);
                }
                rb[c].x = w[0]; rb[c].y = w[1]; rb[c].z = w[2]; rb[c].w = w[3];
            }
        }
    };

    uint4 ra[4], rb[4], na[4], nb[4];
    load_tile(0, ra, rb);

    for (int k0 = 0; k0 < K; k0 += 64) {
        // ---- commit current tile registers to LDS ----
        {
            uint4* da = (uint4*)aDst;
            uint4* db = (uint4*)bDst;
            #pragma unroll
            for (int c = 0; c < 4; ++c) da[c] = ra[c];
            #pragma unroll
            for (int c = 0; c < 4; ++c) db[c] = rb[c];
        }
        __syncthreads();

        // ---- issue next tile's global loads (overlap with WMMA below) ----
        if (k0 + 64 < K) load_tile(k0 + 64, na, nb);

        // ---- two K=32 WMMA steps ----
        #pragma unroll
        for (int kk = 0; kk < 64; kk += 32) {
            v16bf a[2], b[4];
            #pragma unroll
            for (int i = 0; i < 2; ++i) {
                // A layout: lanes 0-15 K={0-7,16-23}, lanes 16-31 K={8-15,24-31}
                const unsigned short* p = &Asl[(wm + 16 * i + l16) * LDSS + kk + 8 * hlf];
                union { v16bf v; uint4 q[2]; } u;
                u.q[0] = *(const uint4*)p;
                u.q[1] = *(const uint4*)(p + 16);
                a[i] = u.v;
            }
            #pragma unroll
            for (int j = 0; j < 4; ++j) {
                // B layout: lane = N col; lanes 0-15 K={0-7,16-23}, lanes 16-31 K={8-15,24-31}
                const unsigned short* p = &Bsl[(wn + 16 * j + l16) * LDSS + kk + 8 * hlf];
                union { v16bf v; uint4 q[2]; } u;
                u.q[0] = *(const uint4*)p;
                u.q[1] = *(const uint4*)(p + 16);
                b[j] = u.v;
            }
            #pragma unroll
            for (int i = 0; i < 2; ++i)
                #pragma unroll
                for (int j = 0; j < 4; ++j)
                    acc[i][j] = __builtin_amdgcn_wmma_f32_16x16x32_bf16(
                        false, a[i], false, b[j], (short)0, acc[i][j], false, false);
        }
        __syncthreads();

        // ---- rotate pipeline registers ----
        #pragma unroll
        for (int c = 0; c < 4; ++c) { ra[c] = na[c]; rb[c] = nb[c]; }
    }

    const float sc = (epi == 3 && scale_ptr) ? scale_ptr[0] : 1.0f;

    // ---- epilogue: C/D layout: n = lane&15 (+16j), m = r + 8*(lane>>4) (+16i)
    #pragma unroll
    for (int i = 0; i < 2; ++i) {
        #pragma unroll
        for (int j = 0; j < 4; ++j) {
            int n = n0 + wn + 16 * j + l16;
            if (n >= N) continue;
            float bv = bias ? bias[n] : 0.f;
            #pragma unroll
            for (int r = 0; r < 8; ++r) {
                int m = m0 + wm + 16 * i + 8 * hlf + r;
                if (m >= M) continue;
                float v = acc[i][j][r] + bv;
                if      (epi == 1) v = fmaxf(v, 0.f);
                else if (epi == 2) v = fmaxf(v * aux[(size_t)m * ldAux + n], 0.f);
                else if (epi == 3) v = fmaxf(v * sc, 0.f);
                size_t ci = (size_t)m * ldC + n;
                if (c_bf16) {
                    ((unsigned short*)Cv)[ci] = f2bf(v);
                } else {
                    float* Cf = (float*)Cv;
                    if (epi == 4) v += Cf[ci];
                    Cf[ci] = v;
                }
            }
        }
    }
}

// ---------------------------------------------------------------------------
// helpers
// ---------------------------------------------------------------------------
__global__ void conv_bf16_kernel(const float* __restrict__ src,
                                 unsigned short* __restrict__ dst, size_t n) {
    size_t i = ((size_t)blockIdx.x * 256 + threadIdx.x) * 4;
    #pragma unroll
    for (int e = 0; e < 4; ++e)
        if (i + e < n) dst[i + e] = f2bf(src[i + e]);
}

// dst[c][r] = src[r][c0+c]  (f32 -> bf16, transposed)  dst leading dim = ldD
__global__ __launch_bounds__(256) void tconv_kernel(
    const float* __restrict__ src, int ldS, int c0,
    unsigned short* __restrict__ dst, int ldD, int R, int C)
{
    __shared__ float tile[32][33];
    int tx = threadIdx.x, ty = threadIdx.y;
    int rb = blockIdx.y * 32, cb = blockIdx.x * 32;
    #pragma unroll
    for (int i = 0; i < 4; ++i) {
        int r = rb + ty + 8 * i, c = cb + tx;
        tile[ty + 8 * i][tx] = (r < R && c < C) ? src[(size_t)r * ldS + c0 + c] : 0.f;
    }
    __syncthreads();
    #pragma unroll
    for (int i = 0; i < 4; ++i) {
        int c = cb + ty + 8 * i, r = rb + tx;
        if (c < C && r < R) dst[(size_t)c * ldD + r] = f2bf(tile[tx][ty + 8 * i]);
    }
}

__global__ void gather_prod_kernel(const float* __restrict__ edge_rep,
                                   const int* __restrict__ pairs,
                                   unsigned short* __restrict__ prod) {
    size_t idx = (size_t)blockIdx.x * 256 + threadIdx.x;   // N_RELP*1024 total
    int r = (int)(idx >> 10), j = (int)(idx & 1023);
    int src = (j < HID) ? pairs[2 * r] : pairs[2 * r + 1];
    prod[idx] = f2bf(edge_rep[(size_t)src * (2 * HID) + j]);
}

__global__ __launch_bounds__(256) void colsum_kernel(const float* __restrict__ tmp,
                                                     float* __restrict__ partial) {
    __shared__ float s[256];
    int col = blockIdx.x;  // 0..199 (u cols then v cols)
    float a = 0.f;
    for (int r = threadIdx.x; r < N_RELP; r += 256) a += tmp[(size_t)r * 400 + col];
    s[threadIdx.x] = a; __syncthreads();
    for (int st = 128; st > 0; st >>= 1) {
        if (threadIdx.x < st) s[threadIdx.x] += s[threadIdx.x + st];
        __syncthreads();
    }
    if (threadIdx.x == 0) partial[col] = s[0];
}

__global__ void dscale_kernel(const float* __restrict__ partial, float* __restrict__ d) {
    __shared__ float s[128];
    float a = 0.f;
    for (int j = threadIdx.x; j < KRANK; j += 128) a += partial[j] * partial[j + KRANK];
    s[threadIdx.x] = a; __syncthreads();
    for (int st = 64; st > 0; st >>= 1) {
        if (threadIdx.x < st) s[threadIdx.x] += s[threadIdx.x + st];
        __syncthreads();
    }
    if (threadIdx.x == 0) d[0] = 1.0f / (s[0] / (float)N_RELP + 1e-6f);
}

__global__ void copy_t_relu_kernel(const float* __restrict__ tmp,
                                   unsigned short* __restrict__ xcat) {
    size_t idx = (size_t)blockIdx.x * 256 + threadIdx.x;   // N_RELP*KRANK
    if (idx >= (size_t)N_RELP * KRANK) return;
    int r = (int)(idx / KRANK), j = (int)(idx % KRANK);
    xcat[(size_t)r * CATW + KRANK + j] = f2bf(fmaxf(tmp[(size_t)r * 400 + 300 + j], 0.f));
}

// x = GN(y) (y already relu'd); writes x (bf16) and relu(x) into xcat cols 200+
__global__ __launch_bounds__(256) void groupnorm_kernel(
    const float* __restrict__ y, const float* __restrict__ gamma,
    const float* __restrict__ beta, unsigned short* __restrict__ xbf,
    unsigned short* __restrict__ xcat)
{
    __shared__ float ssum[256], ssq[256];
    int row = blockIdx.x >> 3, g = blockIdx.x & 7;
    size_t base = (size_t)row * POOL + g * 512;
    float v0 = y[base + threadIdx.x];
    float v1 = y[base + 256 + threadIdx.x];
    ssum[threadIdx.x] = v0 + v1;
    ssq[threadIdx.x]  = v0 * v0 + v1 * v1;
    __syncthreads();
    for (int st = 128; st > 0; st >>= 1) {
        if (threadIdx.x < st) {
            ssum[threadIdx.x] += ssum[threadIdx.x + st];
            ssq[threadIdx.x]  += ssq[threadIdx.x + st];
        }
        __syncthreads();
    }
    float mu = ssum[0] * (1.0f / 512.0f);
    float var = ssq[0] * (1.0f / 512.0f) - mu * mu;
    float rs = rsqrtf(var + 1e-5f);
    #pragma unroll
    for (int e = 0; e < 2; ++e) {
        int cidx = g * 512 + e * 256 + threadIdx.x;
        float v = e ? v1 : v0;
        float xn = (v - mu) * rs * gamma[cidx] + beta[cidx];
        xbf[(size_t)row * POOL + cidx] = f2bf(xn);
        xcat[(size_t)row * CATW + 200 + cidx] = f2bf(fmaxf(xn, 0.f));
    }
}

__global__ void freq_add_kernel(float* __restrict__ out, const float* __restrict__ freq,
                                const int* __restrict__ pairs,
                                const int* __restrict__ obj_preds) {
    int r = blockIdx.x, c = threadIdx.x;
    if (c < NREL) {
        int p0 = obj_preds[pairs[2 * r]];
        int p1 = obj_preds[pairs[2 * r + 1]];
        out[(size_t)r * NREL + c] += freq[((size_t)p0 * NOBJC + p1) * NREL + c];
    }
}

// ---------------------------------------------------------------------------
// LRGA block: tmp = relu(A@Wa+b); d = 1/(su.sv/M+eps); res = u@(v^T z);
// xcat[:,0:100] = relu(res*d); xcat[:,100:200] = relu(t)
// ---------------------------------------------------------------------------
static void enqueue_lrga(hipStream_t stream,
                         const unsigned short* A_bf, const unsigned short* wT_at,
                         const float* b_attn, float* tmp, unsigned short* tmp_bf,
                         unsigned short* vT, unsigned short* zT, float* vTz,
                         unsigned short* vTzT, float* partial, float* dsc,
                         unsigned short* xcat)
{
    gemm_bf16_wmma<<<dim3((400+127)/128, N_RELP/128), 256, 0, stream>>>(
        A_bf, POOL, wT_at, POOL, tmp, 400, 0, N_RELP, 400, POOL,
        b_attn, 1, nullptr, 0, nullptr);
    conv_bf16_kernel<<<(unsigned)(((size_t)N_RELP*400/4 + 255)/256), 256, 0, stream>>>(
        tmp, tmp_bf, (size_t)N_RELP * 400);
    tconv_kernel<<<dim3(4, N_RELP/32), dim3(32,8), 0, stream>>>(tmp, 400, 100, vT, N_RELP, N_RELP, KRANK);
    tconv_kernel<<<dim3(4, N_RELP/32), dim3(32,8), 0, stream>>>(tmp, 400, 200, zT, N_RELP, N_RELP, KRANK);
    colsum_kernel<<<200, 256, 0, stream>>>(tmp, partial);
    dscale_kernel<<<1, 128, 0, stream>>>(partial, dsc);
    gemm_bf16_wmma<<<dim3(1,1), 256, 0, stream>>>(
        vT, N_RELP, zT, N_RELP, vTz, KRANK, 0, KRANK, KRANK, N_RELP,
        nullptr, 0, nullptr, 0, nullptr);
    tconv_kernel<<<dim3(4,4), dim3(32,8), 0, stream>>>(vTz, KRANK, 0, vTzT, KRANK, KRANK, KRANK);
    gemm_bf16_wmma<<<dim3(1, N_RELP/128), 256, 0, stream>>>(
        tmp_bf, 400, vTzT, KRANK, xcat, CATW, 1, N_RELP, KRANK, KRANK,
        nullptr, 3, nullptr, 0, dsc);
    copy_t_relu_kernel<<<(unsigned)(((size_t)N_RELP*KRANK + 255)/256), 256, 0, stream>>>(tmp, xcat);
}

// ---------------------------------------------------------------------------
extern "C" void kernel_launch(void* const* d_in, const int* in_sizes, int n_in,
                              void* d_out, int out_size, void* d_ws, size_t ws_size,
                              hipStream_t stream) {
    (void)in_sizes; (void)n_in; (void)out_size; (void)ws_size;
    const float* edge_ctx  = (const float*)d_in[0];
    const float* unionf    = (const float*)d_in[1];
    const int*   pairs     = (const int*)d_in[2];
    const int*   obj_preds = (const int*)d_in[3];
    const float* w_pe  = (const float*)d_in[4];  const float* b_pe  = (const float*)d_in[5];
    const float* w_pc  = (const float*)d_in[6];  const float* b_pc  = (const float*)d_in[7];
    const float* w_a1  = (const float*)d_in[8];  const float* b_a1  = (const float*)d_in[9];
    const float* w_d1  = (const float*)d_in[10]; const float* b_d1  = (const float*)d_in[11];
    const float* gng   = (const float*)d_in[12]; const float* gnb   = (const float*)d_in[13];
    const float* w_a2  = (const float*)d_in[14]; const float* b_a2  = (const float*)d_in[15];
    const float* w_d2  = (const float*)d_in[16]; const float* b_d2  = (const float*)d_in[17];
    const float* w_rel = (const float*)d_in[18]; const float* b_rel = (const float*)d_in[19];
    const float* w_ctx = (const float*)d_in[20]; const float* b_ctx = (const float*)d_in[21];
    const float* freq  = (const float*)d_in[22];
    float* out = (float*)d_out;

    char* ws = (char*)d_ws;
    size_t off = 0;
    auto take = [&](size_t bytes) -> char* {
        char* p = ws + off; off += (bytes + 255) & ~(size_t)255; return p;
    };
    unsigned short* edge_bf = (unsigned short*)take((size_t)N_OBJ*HID*2);
    unsigned short* wT_pe   = (unsigned short*)take((size_t)1024*HID*2);
    float*          edge_rep= (float*)take((size_t)N_OBJ*1024*4);
    unsigned short* prod_bf = (unsigned short*)take((size_t)N_RELP*1024*2);
    unsigned short* wT_pc   = (unsigned short*)take((size_t)POOL*1024*2);
    unsigned short* big_bf  = (unsigned short*)take((size_t)N_RELP*POOL*2);   // union->x->visual2
    unsigned short* wT_at   = (unsigned short*)take((size_t)400*POOL*2);      // attn1 then attn2
    float*          tmp     = (float*)take((size_t)N_RELP*400*4);
    unsigned short* tmp_bf  = (unsigned short*)take((size_t)N_RELP*400*2);
    unsigned short* vT      = (unsigned short*)take((size_t)KRANK*N_RELP*2);
    unsigned short* zT      = (unsigned short*)take((size_t)KRANK*N_RELP*2);
    float*          vTz     = (float*)take((size_t)KRANK*KRANK*4);
    unsigned short* vTzT    = (unsigned short*)take((size_t)KRANK*KRANK*2);
    float*          partial = (float*)take(200*4);
    float*          dsc     = (float*)take(256);
    unsigned short* xcat    = (unsigned short*)take((size_t)N_RELP*CATW*2);   // xcat1 then xcat2
    unsigned short* wT_dr   = (unsigned short*)take((size_t)POOL*CATW*2);     // dr1 then dr2
    float*          ybuf    = (float*)take((size_t)N_RELP*POOL*4);
    unsigned short* wT_rel  = (unsigned short*)take((size_t)NREL*POOL*2);
    unsigned short* wT_ctx  = (unsigned short*)take((size_t)NREL*1024*2);

    // 1. edge_rep = edge_ctx @ w_post_emb + b
    conv_bf16_kernel<<<(unsigned)(((size_t)N_OBJ*HID/4 + 255)/256), 256, 0, stream>>>(
        edge_ctx, edge_bf, (size_t)N_OBJ*HID);
    tconv_kernel<<<dim3(1024/32, HID/32), dim3(32,8), 0, stream>>>(w_pe, 1024, 0, wT_pe, HID, HID, 1024);
    gemm_bf16_wmma<<<dim3(1024/128, N_OBJ/128), 256, 0, stream>>>(
        edge_bf, HID, wT_pe, HID, edge_rep, 1024, 0, N_OBJ, 1024, HID,
        b_pe, 0, nullptr, 0, nullptr);

    // 2. prod_rep gather (bf16)
    gather_prod_kernel<<<(unsigned)((size_t)N_RELP*1024/256), 256, 0, stream>>>(edge_rep, pairs, prod_bf);

    // 3. transposed-bf16 weights, union bf16
    tconv_kernel<<<dim3(POOL/32, 1024/32), dim3(32,8), 0, stream>>>(w_pc, POOL, 0, wT_pc, 1024, 1024, POOL);
    conv_bf16_kernel<<<(unsigned)(((size_t)N_RELP*POOL/4 + 255)/256), 256, 0, stream>>>(
        unionf, big_bf, (size_t)N_RELP*POOL);
    tconv_kernel<<<dim3((400+31)/32, POOL/32), dim3(32,8), 0, stream>>>(w_a1, 400, 0, wT_at, POOL, POOL, 400);

    // 4. LRGA #1 on union -> xcat[:,0:200]
    enqueue_lrga(stream, big_bf, wT_at, b_a1, tmp, tmp_bf, vT, zT, vTz, vTzT, partial, dsc, xcat);

    // 5. x_local = relu(union * (prod@w_pc + b)) -> xcat[:,200:]
    gemm_bf16_wmma<<<dim3(POOL/128, N_RELP/128), 256, 0, stream>>>(
        prod_bf, 1024, wT_pc, 1024, xcat + 200, CATW, 1, N_RELP, POOL, 1024,
        b_pc, 2, unionf, POOL, nullptr);

    // 6. y = relu(xcat @ w_dr1 + b); x = GN(y) -> big_bf + xcat[:,200:]
    tconv_kernel<<<dim3(POOL/32, (CATW+31)/32), dim3(32,8), 0, stream>>>(w_d1, POOL, 0, wT_dr, CATW, CATW, POOL);
    gemm_bf16_wmma<<<dim3(POOL/128, N_RELP/128), 256, 0, stream>>>(
        xcat, CATW, wT_dr, CATW, ybuf, POOL, 0, N_RELP, POOL, CATW,
        b_d1, 1, nullptr, 0, nullptr);
    groupnorm_kernel<<<N_RELP*8, 256, 0, stream>>>(ybuf, gng, gnb, big_bf, xcat);

    // 7. LRGA #2 on x -> xcat[:,0:200]
    tconv_kernel<<<dim3((400+31)/32, POOL/32), dim3(32,8), 0, stream>>>(w_a2, 400, 0, wT_at, POOL, POOL, 400);
    enqueue_lrga(stream, big_bf, wT_at, b_a2, tmp, tmp_bf, vT, zT, vTz, vTzT, partial, dsc, xcat);

    // 8. visual2 = xcat @ w_dr2 + b (bf16, reuses big_bf)
    tconv_kernel<<<dim3(POOL/32, (CATW+31)/32), dim3(32,8), 0, stream>>>(w_d2, POOL, 0, wT_dr, CATW, CATW, POOL);
    gemm_bf16_wmma<<<dim3(POOL/128, N_RELP/128), 256, 0, stream>>>(
        xcat, CATW, wT_dr, CATW, big_bf, POOL, 1, N_RELP, POOL, CATW,
        b_d2, 0, nullptr, 0, nullptr);

    // 9. rel_dists = visual2@w_rel + b_rel + prod@w_ctx + b_ctx + freq
    tconv_kernel<<<dim3((NREL+31)/32, POOL/32), dim3(32,8), 0, stream>>>(w_rel, NREL, 0, wT_rel, POOL, POOL, NREL);
    tconv_kernel<<<dim3((NREL+31)/32, 1024/32), dim3(32,8), 0, stream>>>(w_ctx, NREL, 0, wT_ctx, 1024, 1024, NREL);
    gemm_bf16_wmma<<<dim3(1, N_RELP/128), 256, 0, stream>>>(
        big_bf, POOL, wT_rel, POOL, out, NREL, 0, N_RELP, NREL, POOL,
        b_rel, 0, nullptr, 0, nullptr);
    gemm_bf16_wmma<<<dim3(1, N_RELP/128), 256, 0, stream>>>(
        prod_bf, 1024, wT_ctx, 1024, out, NREL, 0, N_RELP, NREL, 1024,
        b_ctx, 4, nullptr, 0, nullptr);
    freq_add_kernel<<<N_RELP, 64, 0, stream>>>(out, freq, pairs, obj_preds);
}
